// SpatialBottleneck_9689446220620
// MI455X (gfx1250) — matrix operand order repaired and checked by reading, not compiled, not run on hardware
//
#include <hip/hip_runtime.h>

// ---------------------------------------------------------------------------
// CDNA5 / gfx1250 implicit-GEMM bottleneck:
//   conv1 1x1 (1024->256) + BN + ReLU
//   conv2 3x3 pad1 (256->256) + BN + ReLU   (im2col on the fly)
//   conv3 1x1 (256->1024) + BN + residual + ReLU
// bf16 WMMA (v_wmma_f32_16x16x32_bf16), f32 accumulate, fused epilogues.
// Double-buffered LDS: A-tiles staged with GLOBAL_LOAD_ASYNC_TO_LDS_B128
// one K-step ahead (ASYNCcnt), B-tiles scatter-transposed under compute.
// ---------------------------------------------------------------------------

typedef __attribute__((ext_vector_type(16))) __bf16 v16bf;
typedef __attribute__((ext_vector_type(8)))  float  v8f;

typedef unsigned short u16;
typedef unsigned int   u32;
typedef int v4i __attribute__((vector_size(16)));  // matches builtin param

#if defined(__has_builtin)
#if __has_builtin(__builtin_amdgcn_global_load_async_to_lds_b128) && \
    __has_builtin(__builtin_amdgcn_s_wait_asynccnt)
#define HAVE_ASYNC_LDS 1
#endif
#endif

#define GLB_CAST(p) ((__attribute__((address_space(1))) v4i*)(p))
#define LDS_CAST(p) ((__attribute__((address_space(3))) v4i*)(p))

union FragBF {
  v16bf bf;
  uint4 q[2];
};

union Pix16 {
  uint4 q[2];
  u16   e[16];
};

__device__ __forceinline__ u16 f2bf(float f) {
  // round-to-nearest-even f32 -> bf16
  u32 u = __float_as_uint(f);
  u32 r = (u + 0x7FFFu + ((u >> 16) & 1u)) >> 16;
  return (u16)r;
}

// Problem dims
#define BATCH 32
#define CIN   1024
#define CB    256
#define HH    28
#define HW    784                 // 28*28; 784 = 49*16 -> 16-aligned pixel
#define NPIX  (BATCH * HW)        //   groups never cross a sample boundary
#define XSTR  ((size_t)CIN * HW)  // per-sample stride of x / out
#define OSTR  ((size_t)CB  * HW)  // per-sample stride of intermediates

// Tiling
#define BM  64
#define BN  128
#define BK  32
#define LDA 48   // padded K-stride (ushorts) in LDS A tile
#define LDB 48   // padded K-stride (ushorts) in LDS B tile

// ---------------------------------------------------------------------------
// vectorized fp32 -> bf16 conversion (all sizes used are multiples of 4)
// ---------------------------------------------------------------------------
__global__ __launch_bounds__(256) void cvt4_bf16_kernel(
    const float4* __restrict__ src, ushort4* __restrict__ dst, int n4) {
  int i = blockIdx.x * 256 + threadIdx.x;
  if (i < n4) {
    float4 f = src[i];
    ushort4 o;
    o.x = f2bf(f.x);
    o.y = f2bf(f.y);
    o.z = f2bf(f.z);
    o.w = f2bf(f.w);
    dst[i] = o;
  }
}

// ---------------------------------------------------------------------------
// stage one A (weights) b128 chunk: async to LDS if available
// ---------------------------------------------------------------------------
__device__ __forceinline__ void stage_a_b128(const u16* gsrc, u16* ldst) {
#ifdef HAVE_ASYNC_LDS
  __builtin_amdgcn_global_load_async_to_lds_b128(GLB_CAST(gsrc),
                                                 LDS_CAST(ldst), 0, 0);
#else
  *(uint4*)ldst = *(const uint4*)gsrc;
#endif
}

__device__ __forceinline__ void wait_stage() {
#ifdef HAVE_ASYNC_LDS
  __builtin_amdgcn_s_wait_asynccnt(0);
#endif
}

// fragment A load (lane layout of V_WMMA_*_16X16X32 A operand)
__device__ __forceinline__ FragBF load_a(const u16* Ar, int kg) {
  FragBF a;
  a.q[0] = *(const uint4*)(Ar + kg);
  a.q[1] = *(const uint4*)(Ar + kg + 16);
  return a;
}
// fragment B load (lane layout of V_WMMA_*_16X16X32 B operand)
__device__ __forceinline__ FragBF load_b(const u16* Bc, int kb) {
  FragBF b;
  b.q[0] = *(const uint4*)(Bc + kb);
  b.q[1] = *(const uint4*)(Bc + kb + 16);
  return b;
}

// ---------------------------------------------------------------------------
// conv1: out1 = relu(s1 * (W1 x) + b1); W1: 256x1024 bf16, x: bf16 1024xNPIX
// ---------------------------------------------------------------------------
__global__ __launch_bounds__(256) void conv1_kernel(
    const u16* __restrict__ xb, const u16* __restrict__ w1b,
    const float* __restrict__ s1, const float* __restrict__ b1,
    u16* __restrict__ out1b) {
  __shared__ __align__(16) u16 As[2][BM * LDA];
  __shared__ __align__(16) u16 Bs[2][BN * LDB];

  const int t     = threadIdx.x;
  const int tileM = blockIdx.y * BM;
  const int tileN = blockIdx.x * BN;

  const int lane = t & 31;
  const int wave = t >> 5;
  const int wm   = wave & 3;   // wave M row (0..3)
  const int wn   = wave >> 2;  // wave N col (0..1)
  const int hl   = lane >> 4;  // lane half
  const int l16  = lane & 15;

  // cooperative-load mappings (hoisted out of the K loop)
  const int am  = t >> 2;       // A: row within tile
  const int ak  = (t & 3) * 8;  // A: k offset (8 bf16 = one b128)
  const int kL  = t >> 3;       // B: k lane within tile (0..31)
  const int nb0 = (t & 7) * 16; // B: 16-pixel group base

  // B-source: 16-aligned pixel group -> single sample, computed once
  const int  p0   = tileN + nb0;
  const int  n0   = p0 / HW;
  const int  hw0  = p0 - n0 * HW;
  const u16* bsrc = xb + (size_t)n0 * XSTR + (size_t)kL * HW + hw0;
  const u16* asrc = w1b + (size_t)(tileM + am) * CIN + ak;

  auto stage = [&](int k0, int buf) {
    stage_a_b128(asrc + k0, &As[buf][am * LDA + ak]);
    const u16* s = bsrc + (size_t)k0 * HW;
    Pix16 tv;
    tv.q[0] = *(const uint4*)(s);
    tv.q[1] = *(const uint4*)(s + 8);
    #pragma unroll
    for (int i = 0; i < 16; ++i) Bs[buf][(nb0 + i) * LDB + kL] = tv.e[i];
  };

  v8f acc[4] = {};
  const int STEPS = CIN / BK;
  stage(0, 0);
  for (int i = 0; i < STEPS; ++i) {
    const int cur = i & 1;
    wait_stage();
    __syncthreads();
    if (i + 1 < STEPS) stage((i + 1) * BK, cur ^ 1);

    const u16* Ar = &As[cur][(wm * 16 + l16) * LDA];
    const u16* Bb = &Bs[cur][(wn * 64 + l16) * LDB];
    const int  kg = hl * 8, kb = hl * 8;
    FragBF a  = load_a(Ar, kg);
    FragBF b0 = load_b(Bb + 0 * 16 * LDB, kb);
    FragBF b1 = load_b(Bb + 1 * 16 * LDB, kb);
    FragBF b2 = load_b(Bb + 2 * 16 * LDB, kb);
    FragBF b3 = load_b(Bb + 3 * 16 * LDB, kb);
    acc[0] = __builtin_amdgcn_wmma_f32_16x16x32_bf16(false, a.bf, false, b0.bf,
                                                     (short)0, acc[0], false,
                                                     false);
    acc[1] = __builtin_amdgcn_wmma_f32_16x16x32_bf16(false, a.bf, false, b1.bf,
                                                     (short)0, acc[1], false,
                                                     false);
    acc[2] = __builtin_amdgcn_wmma_f32_16x16x32_bf16(false, a.bf, false, b2.bf,
                                                     (short)0, acc[2], false,
                                                     false);
    acc[3] = __builtin_amdgcn_wmma_f32_16x16x32_bf16(false, a.bf, false, b3.bf,
                                                     (short)0, acc[3], false,
                                                     false);
  }

  const int coB = tileM + wm * 16 + hl * 8;
  float sc[8], bi[8];
  #pragma unroll
  for (int r = 0; r < 8; ++r) {
    sc[r] = s1[coB + r];
    bi[r] = b1[coB + r];
  }
  #pragma unroll
  for (int j = 0; j < 4; ++j) {
    const int p0j = tileN + wn * 64 + j * 16;
    const int n   = p0j / HW;
    const int hw  = p0j - n * HW + l16;
    u16* dst = out1b + (size_t)n * OSTR + (size_t)coB * HW + hw;
    #pragma unroll
    for (int r = 0; r < 8; ++r) {
      float o = fmaxf(acc[j][r] * sc[r] + bi[r], 0.0f);
      dst[(size_t)r * HW] = f2bf(o);
    }
  }
}

// ---------------------------------------------------------------------------
// conv2: 3x3 pad1, implicit GEMM, K = 256*9 = 2304 (im2col on the fly)
// ---------------------------------------------------------------------------
__global__ __launch_bounds__(256) void conv2_kernel(
    const u16* __restrict__ out1b, const u16* __restrict__ w2b,
    const float* __restrict__ s2, const float* __restrict__ b2,
    u16* __restrict__ out2b) {
  __shared__ __align__(16) u16 As[2][BM * LDA];
  __shared__ __align__(16) u16 Bs[2][BN * LDB];

  const int t     = threadIdx.x;
  const int tileM = blockIdx.y * BM;
  const int tileN = blockIdx.x * BN;

  const int lane = t & 31;
  const int wave = t >> 5;
  const int wm   = wave & 3;
  const int wn   = wave >> 2;
  const int hl   = lane >> 4;
  const int l16  = lane & 15;

  const int am  = t >> 2;
  const int ak  = (t & 3) * 8;
  const int kL  = t >> 3;
  const int nb0 = (t & 7) * 16;

  const int KTOT = CB * 9;  // 2304

  // B-source pixel group geometry (hoisted)
  const int  p0  = tileN + nb0;
  const int  n0  = p0 / HW;
  const int  hw0 = p0 - n0 * HW;
  const int  h0  = hw0 / HH;
  const int  w0  = hw0 - h0 * HH;
  const u16* bbase = out1b + (size_t)n0 * OSTR;
  const u16* asrc  = w2b + (size_t)(tileM + am) * KTOT + ak;

  auto stage = [&](int k0, int buf) {
    stage_a_b128(asrc + k0, &As[buf][am * LDA + ak]);
    const int kk = k0 + kL;
    const int ci = kk / 9;
    const int rr = kk - ci * 9;
    const int dy = rr / 3 - 1;
    const int dx = rr - (rr / 3) * 3 - 1;
    const u16* src = bbase + (size_t)ci * HW;
    int hb = h0, wb = w0;
    #pragma unroll
    for (int i = 0; i < 16; ++i) {
      int h = hb + dy;
      int w = wb + dx;
      u16 v = 0;
      if (h >= 0 && h < HH && w >= 0 && w < HH) v = src[h * HH + w];
      Bs[buf][(nb0 + i) * LDB + kL] = v;
      if (++wb == HH) { wb = 0; ++hb; }
    }
  };

  v8f acc[4] = {};
  const int STEPS = KTOT / BK;
  stage(0, 0);
  for (int i = 0; i < STEPS; ++i) {
    const int cur = i & 1;
    wait_stage();
    __syncthreads();
    if (i + 1 < STEPS) stage((i + 1) * BK, cur ^ 1);

    const u16* Ar = &As[cur][(wm * 16 + l16) * LDA];
    const u16* Bb = &Bs[cur][(wn * 64 + l16) * LDB];
    const int  kg = hl * 8, kb = hl * 8;
    FragBF a  = load_a(Ar, kg);
    FragBF b0 = load_b(Bb + 0 * 16 * LDB, kb);
    FragBF b1 = load_b(Bb + 1 * 16 * LDB, kb);
    FragBF b2 = load_b(Bb + 2 * 16 * LDB, kb);
    FragBF b3 = load_b(Bb + 3 * 16 * LDB, kb);
    acc[0] = __builtin_amdgcn_wmma_f32_16x16x32_bf16(false, a.bf, false, b0.bf,
                                                     (short)0, acc[0], false,
                                                     false);
    acc[1] = __builtin_amdgcn_wmma_f32_16x16x32_bf16(false, a.bf, false, b1.bf,
                                                     (short)0, acc[1], false,
                                                     false);
    acc[2] = __builtin_amdgcn_wmma_f32_16x16x32_bf16(false, a.bf, false, b2.bf,
                                                     (short)0, acc[2], false,
                                                     false);
    acc[3] = __builtin_amdgcn_wmma_f32_16x16x32_bf16(false, a.bf, false, b3.bf,
                                                     (short)0, acc[3], false,
                                                     false);
  }

  const int coB = tileM + wm * 16 + hl * 8;
  float sc[8], bi[8];
  #pragma unroll
  for (int r = 0; r < 8; ++r) {
    sc[r] = s2[coB + r];
    bi[r] = b2[coB + r];
  }
  #pragma unroll
  for (int j = 0; j < 4; ++j) {
    const int p0j = tileN + wn * 64 + j * 16;
    const int n   = p0j / HW;
    const int hw  = p0j - n * HW + l16;
    u16* dst = out2b + (size_t)n * OSTR + (size_t)coB * HW + hw;
    #pragma unroll
    for (int r = 0; r < 8; ++r) {
      float o = fmaxf(acc[j][r] * sc[r] + bi[r], 0.0f);
      dst[(size_t)r * HW] = f2bf(o);
    }
  }
}

// ---------------------------------------------------------------------------
// conv3: 1x1 (256->1024) + BN + residual(x) + ReLU, f32 output
// ---------------------------------------------------------------------------
__global__ __launch_bounds__(256) void conv3_kernel(
    const u16* __restrict__ out2b, const u16* __restrict__ w3b,
    const float* __restrict__ s3, const float* __restrict__ b3,
    const float* __restrict__ x, float* __restrict__ out) {
  __shared__ __align__(16) u16 As[2][BM * LDA];
  __shared__ __align__(16) u16 Bs[2][BN * LDB];

  const int t     = threadIdx.x;
  const int tileM = blockIdx.y * BM;
  const int tileN = blockIdx.x * BN;

  const int lane = t & 31;
  const int wave = t >> 5;
  const int wm   = wave & 3;
  const int wn   = wave >> 2;
  const int hl   = lane >> 4;
  const int l16  = lane & 15;

  const int am  = t >> 2;
  const int ak  = (t & 3) * 8;
  const int kL  = t >> 3;
  const int nb0 = (t & 7) * 16;

  const int  p0   = tileN + nb0;
  const int  n0   = p0 / HW;
  const int  hw0  = p0 - n0 * HW;
  const u16* bsrc = out2b + (size_t)n0 * OSTR + (size_t)kL * HW + hw0;
  const u16* asrc = w3b + (size_t)(tileM + am) * CB + ak;

  auto stage = [&](int k0, int buf) {
    stage_a_b128(asrc + k0, &As[buf][am * LDA + ak]);
    const u16* s = bsrc + (size_t)k0 * HW;
    Pix16 tv;
    tv.q[0] = *(const uint4*)(s);
    tv.q[1] = *(const uint4*)(s + 8);
    #pragma unroll
    for (int i = 0; i < 16; ++i) Bs[buf][(nb0 + i) * LDB + kL] = tv.e[i];
  };

  v8f acc[4] = {};
  const int STEPS = CB / BK;
  stage(0, 0);
  for (int i = 0; i < STEPS; ++i) {
    const int cur = i & 1;
    wait_stage();
    __syncthreads();
    if (i + 1 < STEPS) stage((i + 1) * BK, cur ^ 1);

    const u16* Ar = &As[cur][(wm * 16 + l16) * LDA];
    const u16* Bb = &Bs[cur][(wn * 64 + l16) * LDB];
    const int  kg = hl * 8, kb = hl * 8;
    FragBF a  = load_a(Ar, kg);
    FragBF b0 = load_b(Bb + 0 * 16 * LDB, kb);
    FragBF b1 = load_b(Bb + 1 * 16 * LDB, kb);
    FragBF b2 = load_b(Bb + 2 * 16 * LDB, kb);
    FragBF b3 = load_b(Bb + 3 * 16 * LDB, kb);
    acc[0] = __builtin_amdgcn_wmma_f32_16x16x32_bf16(false, a.bf, false, b0.bf,
                                                     (short)0, acc[0], false,
                                                     false);
    acc[1] = __builtin_amdgcn_wmma_f32_16x16x32_bf16(false, a.bf, false, b1.bf,
                                                     (short)0, acc[1], false,
                                                     false);
    acc[2] = __builtin_amdgcn_wmma_f32_16x16x32_bf16(false, a.bf, false, b2.bf,
                                                     (short)0, acc[2], false,
                                                     false);
    acc[3] = __builtin_amdgcn_wmma_f32_16x16x32_bf16(false, a.bf, false, b3.bf,
                                                     (short)0, acc[3], false,
                                                     false);
  }

  const int coB = tileM + wm * 16 + hl * 8;
  float sc[8], bi[8];
  #pragma unroll
  for (int r = 0; r < 8; ++r) {
    sc[r] = s3[coB + r];
    bi[r] = b3[coB + r];
  }
  #pragma unroll
  for (int j = 0; j < 4; ++j) {
    const int p0j = tileN + wn * 64 + j * 16;
    const int n   = p0j / HW;
    const int hw  = p0j - n * HW + l16;
    const size_t base = (size_t)n * XSTR + (size_t)coB * HW + hw;
    #pragma unroll
    for (int r = 0; r < 8; ++r) {
      size_t idx = base + (size_t)r * HW;
      float  o   = acc[j][r] * sc[r] + bi[r] + x[idx];
      out[idx]   = fmaxf(o, 0.0f);
    }
  }
}

// ---------------------------------------------------------------------------
// launch
// ---------------------------------------------------------------------------
extern "C" void kernel_launch(void* const* d_in, const int* in_sizes, int n_in,
                              void* d_out, int out_size, void* d_ws,
                              size_t ws_size, hipStream_t stream) {
  const float* x  = (const float*)d_in[0];
  const float* w1 = (const float*)d_in[1];
  const float* w2 = (const float*)d_in[2];
  const float* w3 = (const float*)d_in[3];
  const float* s1 = (const float*)d_in[4];
  const float* b1 = (const float*)d_in[5];
  const float* s2 = (const float*)d_in[6];
  const float* b2 = (const float*)d_in[7];
  const float* s3 = (const float*)d_in[8];
  const float* b3 = (const float*)d_in[9];
  float* out = (float*)d_out;

  // workspace layout (bytes, all 16B aligned)
  char* ws = (char*)d_ws;
  u16* w1b = (u16*)(ws + 0);         //   524288 B (256*1024 bf16)
  u16* w2b = (u16*)(ws + 524288);    //  1179648 B (256*2304 bf16)
  u16* w3b = (u16*)(ws + 1703936);   //   524288 B (1024*256 bf16)
  u16* o1b = (u16*)(ws + 2228224);   // 12845056 B (32*256*784 bf16)
  u16* o2b = (u16*)(ws + 15073280);  // 12845056 B
  u16* xb  = (u16*)(ws + 27918336);  // 51380224 B -> total 79298560 B

  const int nW1 = 256 * 1024, nW2 = 256 * 2304, nW3 = 1024 * 256;
  const int nX  = 32 * 1024 * 784;
  cvt4_bf16_kernel<<<nW1 / 4 / 256, 256, 0, stream>>>(
      (const float4*)w1, (ushort4*)w1b, nW1 / 4);
  cvt4_bf16_kernel<<<nW2 / 4 / 256, 256, 0, stream>>>(
      (const float4*)w2, (ushort4*)w2b, nW2 / 4);
  cvt4_bf16_kernel<<<nW3 / 4 / 256, 256, 0, stream>>>(
      (const float4*)w3, (ushort4*)w3b, nW3 / 4);
  cvt4_bf16_kernel<<<nX / 4 / 256, 256, 0, stream>>>(
      (const float4*)x, (ushort4*)xb, nX / 4);

  // NPIX = 25088 = 196 * BN exactly; M dims divide BM exactly.
  conv1_kernel<<<dim3(196, 4), 256, 0, stream>>>(xb, w1b, s1, b1, o1b);
  conv2_kernel<<<dim3(196, 4), 256, 0, stream>>>(o1b, w2b, s2, b2, o2b);
  conv3_kernel<<<dim3(196, 16), 256, 0, stream>>>(o2b, w3b, s3, b3, x, out);
}
